// OriginalImplementationManifoldLoss_3693671875152
// MI455X (gfx1250) — matrix operand order repaired and checked by reading, not compiled, not run on hardware
//
#include <hip/hip_runtime.h>
#include <hip/hip_bf16.h>
#include <math.h>

// Problem dimensions (fixed by the reference)
#define B1N   512
#define PPN   2048
#define B2N   1536
#define MM    4096            // B1+P+B2
#define DD    512
#define NPROW 2560            // B1+P : rows of the inverse we need
#define KITER 30              // Neumann terms: error <= 2^-30

typedef __attribute__((ext_vector_type(16))) _Float16 v16h;
typedef __attribute__((ext_vector_type(8)))  _Float16 v8h;
typedef __attribute__((ext_vector_type(8)))  float    v8f;

// ---------------------------------------------------------------------------
// WMMA fragment loaders per CDNA5 ISA §7.12.2 (16-bit A 16x32, 16-bit B 32x16)
// A: lane<16 -> row=lh, K {0..7,16..23}; lane>=16 -> row=lh, K {8..15,24..31}
// B: lane<16 -> col=lh, K {0..7,16..23}; lane>=16 -> col=lh, K {8..15,24..31}
//    (we always feed B from row-major rows, via symmetry of S / B = A_p^T)
// ---------------------------------------------------------------------------
__device__ __forceinline__ v16h load_a_frag(const _Float16* base, int ld,
                                            int row0, int kb, int lh, int hi) {
  const _Float16* ap = base + (size_t)(row0 + lh) * ld + kb + (hi << 3);
  v8h a0 = *(const v8h*)ap;         // K +0..7   (or +8..15)
  v8h a1 = *(const v8h*)(ap + 16);  // K +16..23 (or +24..31)
  v16h a;
#pragma unroll
  for (int i = 0; i < 8; ++i) { a[i] = a0[i]; a[i + 8] = a1[i]; }
  return a;
}

__device__ __forceinline__ v16h load_b_frag(const _Float16* base, int ld,
                                            int row0, int kb, int lh, int hi) {
  const _Float16* bp = base + (size_t)(row0 + lh) * ld + kb + (hi << 3);
  v8h b0 = *(const v8h*)bp;         // K +0..7   (or +8..15)
  v8h b1 = *(const v8h*)(bp + 16);  // K +16..23 (or +24..31)
  v16h b;
#pragma unroll
  for (int i = 0; i < 8; ++i) { b[i] = b0[i]; b[i + 8] = b1[i]; }
  return b;
}

// Split-f16 (hi+lo) Gram tile: full ~f32 precision via 3 WMMAs per k-step
__device__ __forceinline__ v8f gram_tile(const _Float16* xhi, const _Float16* xlo,
                                         int it, int jt, int lh, int hi) {
  v8f acc = {};
  for (int kb = 0; kb < DD; kb += 32) {
    v16h ah = load_a_frag(xhi, DD, it * 16, kb, lh, hi);
    v16h al = load_a_frag(xlo, DD, it * 16, kb, lh, hi);
    v16h bh = load_b_frag(xhi, DD, jt * 16, kb, lh, hi);
    v16h bl = load_b_frag(xlo, DD, jt * 16, kb, lh, hi);
    acc = __builtin_amdgcn_wmma_f32_16x16x32_f16(false, al, false, bh, (short)0, acc, false, false);
    acc = __builtin_amdgcn_wmma_f32_16x16x32_f16(false, ah, false, bl, (short)0, acc, false, false);
    acc = __builtin_amdgcn_wmma_f32_16x16x32_f16(false, ah, false, bh, (short)0, acc, false, false);
  }
  return acc;
}

// ---------------------------------------------------------------------------
// 1) concat + row-normalize + split-f16 pack
// ---------------------------------------------------------------------------
__global__ __launch_bounds__(256)
void k_norm_pack(const float* fvec, const float* proxies, const float* fadd,
                 _Float16* xhi, _Float16* xlo) {
  int row = blockIdx.x;
  const float* src;
  if (row < B1N)            src = fvec    + (size_t)row * DD;
  else if (row < B1N + PPN) src = proxies + (size_t)(row - B1N) * DD;
  else                      src = fadd    + (size_t)(row - B1N - PPN) * DD;

  __shared__ float red[256];
  int t = threadIdx.x;
  float v0 = src[t], v1 = src[t + 256];
  red[t] = v0 * v0 + v1 * v1;
  __syncthreads();
  for (int s = 128; s > 0; s >>= 1) {
    if (t < s) red[t] += red[t + s];
    __syncthreads();
  }
  float inv = rsqrtf(red[0]);
  float y0 = v0 * inv, y1 = v1 * inv;
  _Float16 h0 = (_Float16)y0, h1 = (_Float16)y1;
  size_t base = (size_t)row * DD;
  xhi[base + t] = h0;               xhi[base + t + 256] = h1;
  xlo[base + t] = (_Float16)(y0 - (float)h0);
  xlo[base + t + 256] = (_Float16)(y1 - (float)h1);
}

// ---------------------------------------------------------------------------
// 2a) Gram pass A: W row-sums only (per-column-tile partials, no atomics)
// ---------------------------------------------------------------------------
__global__ __launch_bounds__(256)
void k_gram_rowsum(const _Float16* xhi, const _Float16* xlo, float* partial) {
  int wid = (blockIdx.x << 3) + (threadIdx.x >> 5);
  int it = wid >> 8, jt = wid & 255;           // 256 x 256 tiles
  int lane = threadIdx.x & 31;
  int lh = lane & 15, hi = lane >> 4;

  v8f acc = gram_tile(xhi, xlo, it, jt, lh, hi);

  float rs[8];
#pragma unroll
  for (int v = 0; v < 8; ++v) {
    int grow = it * 16 + v + hi * 8;
    int gcol = jt * 16 + lh;
    rs[v] = (grow == gcol) ? 0.f : expf(2.f * fmaxf(acc[v], 0.f));
  }
  // sum over the 16 columns of the tile (xor masks < 16 stay in half-wave)
#pragma unroll
  for (int m = 8; m >= 1; m >>= 1)
#pragma unroll
    for (int v = 0; v < 8; ++v) rs[v] += __shfl_xor(rs[v], m);
  if (lh == 0) {
#pragma unroll
    for (int v = 0; v < 8; ++v)
      partial[(size_t)jt * MM + it * 16 + v + hi * 8] = rs[v];
  }
}

__global__ __launch_bounds__(256)
void k_rowsum_d(const float* partial, float* dvec) {
  int i = blockIdx.x * 256 + threadIdx.x;      // 4096 rows
  float s = 0.f;
  for (int jt = 0; jt < 256; ++jt) s += partial[(size_t)jt * MM + i];
  dvec[i] = (s > 0.f) ? rsqrtf(s) : 0.f;       // isinf(d)->0 guard
}

// ---------------------------------------------------------------------------
// 2b) Gram pass B: build S (f16) and Acc = I + 0.5*S (f32, exact 1st order)
// ---------------------------------------------------------------------------
__global__ __launch_bounds__(256)
void k_gram_S(const _Float16* xhi, const _Float16* xlo, const float* dvec,
              _Float16* Sh, float* Acc) {
  int wid = (blockIdx.x << 3) + (threadIdx.x >> 5);
  int it = wid >> 8, jt = wid & 255;
  int lane = threadIdx.x & 31;
  int lh = lane & 15, hi = lane >> 4;

  v8f acc = gram_tile(xhi, xlo, it, jt, lh, hi);

#pragma unroll
  for (int v = 0; v < 8; ++v) {
    int grow = it * 16 + v + hi * 8;
    int gcol = jt * 16 + lh;
    float w = (grow == gcol) ? 0.f : expf(2.f * fmaxf(acc[v], 0.f));
    float s = dvec[grow] * w * dvec[gcol];
    Sh[(size_t)grow * MM + gcol] = (_Float16)s;
    if (grow < NPROW)
      Acc[(size_t)grow * MM + gcol] = ((grow == gcol) ? 1.f : 0.f) + 0.5f * s;
  }
}

// ---------------------------------------------------------------------------
// 3) Neumann step: Rn = R * S (f16 in, f32 accum), Acc += w * Rn.
//    4x4 register blocking: each wave owns a 64x64 output region.
//    Per k-step: 4 A frags + 4 B frags (8KB) -> 16 WMMAs (256K FLOPs)
//    = 32 FLOPs/byte from L2 (S 32MB + R 20MB are L2-resident; 192MB L2).
//    128-thread blocks (4 waves) guarantee placement at high VGPR usage;
//    16 independent WMMA chains supply the ILP for the 1-wave/SIMD regime.
// ---------------------------------------------------------------------------
__global__ __launch_bounds__(128)
void k_neumann(const _Float16* R, const _Float16* Sh, _Float16* Rn,
               float* Acc, float w) {
  int wid = blockIdx.x * 4 + (threadIdx.x >> 5);
  int itg = wid >> 6;                // 0..39  (2560/64)
  int jtg = wid & 63;                // 0..63  (4096/64)
  int lane = threadIdx.x & 31;
  int lh = lane & 15, hi = lane >> 4;
  int arow = itg * 64;
  int brow = jtg * 64;

  v8f acc[4][4] = {};
  for (int kb = 0; kb < MM; kb += 32) {
    v16h af[4];
#pragma unroll
    for (int i = 0; i < 4; ++i)
      af[i] = load_a_frag(R, MM, arow + i * 16, kb, lh, hi);
#pragma unroll
    for (int j = 0; j < 4; ++j) {
      v16h bf = load_b_frag(Sh, MM, brow + j * 16, kb, lh, hi);  // S symmetric
#pragma unroll
      for (int i = 0; i < 4; ++i)
        acc[i][j] = __builtin_amdgcn_wmma_f32_16x16x32_f16(
            false, af[i], false, bf, (short)0, acc[i][j], false, false);
    }
  }
#pragma unroll
  for (int i = 0; i < 4; ++i) {
#pragma unroll
    for (int j = 0; j < 4; ++j) {
#pragma unroll
      for (int v = 0; v < 8; ++v) {
        int row = arow + i * 16 + v + hi * 8;
        int col = brow + j * 16 + lh;
        size_t idx = (size_t)row * MM + col;
        float x = acc[i][j][v];
        Rn[idx] = (_Float16)x;       // U_{k+1} = S^{k+1} rows (f16-healthy scale)
        Acc[idx] += w * x;           // w = 0.5^{k+1}
      }
    }
  }
}

// ---------------------------------------------------------------------------
// 4) row norms of Acc -> 1/max(0.5*||row||, eps)
// ---------------------------------------------------------------------------
__global__ __launch_bounds__(256)
void k_rnorm(const float* Acc, float* rnorm) {
  int row = blockIdx.x;              // 0..2559
  __shared__ float red[256];
  int t = threadIdx.x;
  float ss = 0.f;
  const float* r = Acc + (size_t)row * MM;
  for (int j = t; j < MM; j += 256) { float x = r[j]; ss += x * x; }
  red[t] = ss;
  __syncthreads();
  for (int s = 128; s > 0; s >>= 1) {
    if (t < s) red[t] += red[t + s];
    __syncthreads();
  }
  rnorm[row] = 1.f / fmaxf(0.5f * sqrtf(red[0]), 1e-8f);
}

// 5) split-f16 pack of Acc (reuses the two R buffers)
__global__ __launch_bounds__(256)
void k_pack(const float* Acc, _Float16* phi, _Float16* plo) {
  size_t idx = (size_t)blockIdx.x * 256 + threadIdx.x;   // 2560*4096 elements
  float x = Acc[idx];
  _Float16 h = (_Float16)x;
  phi[idx] = h;
  plo[idx] = (_Float16)(x - (float)h);
}

// ---------------------------------------------------------------------------
// 6) cosC = 0.25 * (Acc_n . Acc_p^T) * rn * rp  via split-f16 WMMA (~f32 prec)
//    B[k][n] = Acc_p[n][k]  -> lane reads pack row (512 + jt*16 + lh), contiguous
// ---------------------------------------------------------------------------
__global__ __launch_bounds__(256)
void k_cos(const _Float16* phi, const _Float16* plo, const float* rnorm,
           float* cosC) {
  int wid = blockIdx.x * 8 + (threadIdx.x >> 5);
  int it = wid >> 7;                 // 0..31   (512/16)
  int jt = wid & 127;                // 0..127  (2048/16)
  int lane = threadIdx.x & 31;
  int lh = lane & 15, hi = lane >> 4;

  v8f acc = {};
  for (int kb = 0; kb < MM; kb += 32) {
    v16h ah = load_a_frag(phi, MM, it * 16, kb, lh, hi);
    v16h al = load_a_frag(plo, MM, it * 16, kb, lh, hi);
    v16h bh = load_b_frag(phi, MM, B1N + jt * 16, kb, lh, hi);
    v16h bl = load_b_frag(plo, MM, B1N + jt * 16, kb, lh, hi);
    acc = __builtin_amdgcn_wmma_f32_16x16x32_f16(false, al, false, bh, (short)0, acc, false, false);
    acc = __builtin_amdgcn_wmma_f32_16x16x32_f16(false, ah, false, bl, (short)0, acc, false, false);
    acc = __builtin_amdgcn_wmma_f32_16x16x32_f16(false, ah, false, bh, (short)0, acc, false, false);
  }
#pragma unroll
  for (int v = 0; v < 8; ++v) {
    int row = it * 16 + v + hi * 8;  // 0..511
    int colp = jt * 16 + lh;         // 0..2047
    cosC[(size_t)row * PPN + colp] = 0.25f * acc[v] * rnorm[row] * rnorm[B1N + colp];
  }
}

// ---------------------------------------------------------------------------
// 7) per-row loss terms (deterministic fixed-tree block reduce)
// ---------------------------------------------------------------------------
__global__ __launch_bounds__(256)
void k_loss_rows(const float* cosC, const float* Acc, const int* fL,
                 float* row_loss) {
  int i = blockIdx.x;                // 0..511
  int t = threadIdx.x;
  int L = fL[i];
  const float* cr = cosC + (size_t)i * PPN;
  const float* ar = Acc + (size_t)i * MM + B1N;  // Ain row = 0.5*this
  float pos_cos = cr[L];
  float pos_in  = 0.5f * ar[L];
  float sc = 0.f, si = 0.f;
  for (int p = t; p < PPN; p += 256) {
    if (p == L) continue;
    float vc = cr[p] - pos_cos;
    if (vc > 0.f) sc += expf(vc);
    float vi = 0.5f * ar[p] - pos_in;
    if (vi > 0.f) si += expf(vi);
  }
  __shared__ float r1[256], r2[256];
  r1[t] = si; r2[t] = sc;
  __syncthreads();
  for (int s = 128; s > 0; s >>= 1) {
    if (t < s) { r1[t] += r1[t + s]; r2[t] += r2[t + s]; }
    __syncthreads();
  }
  if (t == 0) {
    row_loss[i]        = log1pf(r1[0]);   // intrinsic
    row_loss[B1N + i]  = log1pf(r2[0]);   // context
  }
}

__global__ __launch_bounds__(256)
void k_final(const float* row_loss, float* out) {
  __shared__ float r1[256], r2[256];
  int t = threadIdx.x;
  r1[t] = row_loss[t] + row_loss[t + 256];
  r2[t] = row_loss[B1N + t] + row_loss[B1N + t + 256];
  __syncthreads();
  for (int s = 128; s > 0; s >>= 1) {
    if (t < s) { r1[t] += r1[t + s]; r2[t] += r2[t + s]; }
    __syncthreads();
  }
  if (t == 0) out[0] = r1[0] / (float)B1N + 1.0f * (r2[0] / (float)B1N);
}

// ---------------------------------------------------------------------------
extern "C" void kernel_launch(void* const* d_in, const int* in_sizes, int n_in,
                              void* d_out, int out_size, void* d_ws, size_t ws_size,
                              hipStream_t stream) {
  const float* fvec    = (const float*)d_in[0];
  const int*   fLvec   = (const int*)d_in[1];
  const float* fadd    = (const float*)d_in[2];
  const float* proxies = (const float*)d_in[3];
  float* out = (float*)d_out;

  char* p = (char*)d_ws;
  size_t used = 0;
  auto alloc = [&](size_t bytes) -> void* {
    void* r = p + used;
    used += (bytes + 255) & ~(size_t)255;
    return r;
  };
  _Float16* xhi   = (_Float16*)alloc((size_t)MM * DD * 2);
  _Float16* xlo   = (_Float16*)alloc((size_t)MM * DD * 2);
  _Float16* Sh    = (_Float16*)alloc((size_t)MM * MM * 2);
  float*    partl = (float*)   alloc((size_t)MM * 256 * 4);
  float*    dvec  = (float*)   alloc((size_t)MM * 4);
  float*    Acc   = (float*)   alloc((size_t)NPROW * MM * 4);
  _Float16* R0    = (_Float16*)alloc((size_t)NPROW * MM * 2);
  _Float16* R1    = (_Float16*)alloc((size_t)NPROW * MM * 2);
  float*    rnorm = (float*)   alloc((size_t)NPROW * 4);
  float*    cosC  = (float*)   alloc((size_t)B1N * PPN * 4);
  float*    rloss = (float*)   alloc((size_t)2 * B1N * 4);
  if (used > ws_size) return;  // workspace too small: bail safely

  // 1) normalize + pack
  k_norm_pack<<<MM, 256, 0, stream>>>(fvec, proxies, fadd, xhi, xlo);

  // 2) Gram passes: row sums -> d -> S + Acc init (I + 0.5 S exact in f32)
  k_gram_rowsum<<<(256 * 256) / 8, 256, 0, stream>>>(xhi, xlo, partl);
  k_rowsum_d<<<MM / 256, 256, 0, stream>>>(partl, dvec);
  k_gram_S<<<(256 * 256) / 8, 256, 0, stream>>>(xhi, xlo, dvec, Sh, Acc);

  // 3) Neumann series: U_1 = S (read Sh directly), U_{k+1} = U_k * S
  //    4x4-blocked GEMM: 40x64 wave tiles, 4 waves/block -> 640 blocks
  _Float16* rin = (_Float16*)Sh;   // rows 0..2559 of S, ld = MM
  _Float16* bufs[2] = {R0, R1};
  double w = 0.25;                 // 0.5^2 for the k=2 term
  int cur = 0;
  for (int k = 2; k <= KITER; ++k) {
    _Float16* rout = bufs[cur];
    k_neumann<<<(40 * 64) / 4, 128, 0, stream>>>(rin, Sh, rout, Acc, (float)w);
    rin = rout;
    cur ^= 1;
    w *= 0.5;
  }

  // 4-6) norms, split-pack (reuse R buffers), cosC
  k_rnorm<<<NPROW, 256, 0, stream>>>(Acc, rnorm);
  k_pack<<<(NPROW * MM) / 256, 256, 0, stream>>>(Acc, R0, R1);
  k_cos<<<(32 * 128) / 8, 256, 0, stream>>>(R0, R1, rnorm, cosC);

  // 7) losses
  k_loss_rows<<<B1N, 256, 0, stream>>>(cosC, Acc, fLvec, rloss);
  k_final<<<1, 256, 0, stream>>>(rloss, out);
}
